// QuantizedMambaSSM_69758858821864
// MI455X (gfx1250) — matrix operand order repaired and checked by reading, not compiled, not run on hardware
//
#include <hip/hip_runtime.h>

typedef __attribute__((ext_vector_type(16))) _Float16 v16h;
typedef __attribute__((ext_vector_type(8)))  _Float16 v8h;
typedef __attribute__((ext_vector_type(8)))  float    v8f;
typedef __attribute__((ext_vector_type(4)))  int      v4i;

#define DEV __device__ __forceinline__
#define GLOBAL_AS __attribute__((address_space(1)))
#define LDS_AS    __attribute__((address_space(3)))

#if defined(__has_builtin)
#if __has_builtin(__builtin_amdgcn_global_load_async_to_lds_b128)
#define HAVE_ASYNC_LDS 1
#endif
#if __has_builtin(__builtin_amdgcn_s_wait_asynccnt)
#define HAVE_WAIT_ASYNC 1
#endif
#endif

// Generic->AS1: numerically identical on AMDGPU; bounce through integer so the
// cast is legal C++. Generic->AS3: LDS offset lives in addr[31:0] (ISA aperture
// rules), so truncation to 32 bits is the correct AS3 pointer value.
DEV GLOBAL_AS v4i* gcast_v4i(const void* p) {
  return (GLOBAL_AS v4i*)(unsigned long long)(uintptr_t)p;
}
DEV LDS_AS v4i* lcast_v4i(void* p) {
  return (LDS_AS v4i*)(unsigned)(uintptr_t)p;
}

DEV void wait_async0() {
#if defined(HAVE_WAIT_ASYNC)
  __builtin_amdgcn_s_wait_asynccnt(0);
#else
  asm volatile("s_wait_asynccnt 0x0" ::: "memory");
#endif
}
DEV void wait_async4() {
#if defined(HAVE_WAIT_ASYNC)
  __builtin_amdgcn_s_wait_asynccnt(4);
#else
  asm volatile("s_wait_asynccnt 0x4" ::: "memory");
#endif
}

DEV float softplusf(float x) { return x > 20.f ? x : log1pf(expf(x)); }
DEV float siluf(float x)     { return x / (1.f + expf(-x)); }

DEV v16h combine16(v8h a, v8h b) {
  v16h r;
#pragma unroll
  for (int i = 0; i < 8; ++i) { r[i] = a[i]; r[8 + i] = b[i]; }
  return r;
}

// ---------------------------------------------------------------------------
// Kernel 1: per-row block-Hadamard (FWHT over 12 blocks of 64) + sign1,
// permutation, sign2, int4 fake-quant -> f16 activations for WMMA.
// ---------------------------------------------------------------------------
__global__ void had_quant_kernel(const float* __restrict__ x,
                                 const float* __restrict__ sign1,
                                 const float* __restrict__ sign2,
                                 const int*   __restrict__ perm,
                                 const float* __restrict__ pvals,
                                 _Float16* __restrict__ xq, int D) {
  __shared__ float buf[768];
  __shared__ float buf2[768];
  const int row = blockIdx.x;
  const int tid = threadIdx.x;
  const float* xr = x + (size_t)row * D;
  for (int i = tid; i < D; i += blockDim.x) buf[i] = xr[i];
  __syncthreads();
#pragma unroll
  for (int s = 1; s < 64; s <<= 1) {
    for (int p = tid; p < (768 / 2); p += blockDim.x) {
      int blk = p >> 5, j = p & 31;
      int lo = j & (s - 1);
      int hi = (j & ~(s - 1)) << 1;
      int i0 = blk * 64 + hi + lo;
      int i1 = i0 + s;
      float a = buf[i0], b = buf[i1];
      buf[i0] = a + b;
      buf[i1] = a - b;
    }
    __syncthreads();
  }
  for (int i = tid; i < D; i += blockDim.x) buf2[i] = buf[i] * 0.125f * sign1[i];
  __syncthreads();
  for (int i = tid; i < D; i += blockDim.x) {
    float v  = buf2[perm[i]] * sign2[i];
    float sc = pvals[i] * (1.f / 7.f);
    float q  = rintf(v / (sc + 1e-8f));
    q = fminf(7.f, fmaxf(-7.f, q));
    xq[(size_t)row * D + i] = (_Float16)(q * sc);
  }
}

// ---------------------------------------------------------------------------
// Ternary weight quantization: wq = softplus(s) * sign(w - t) * (|w - t| > d)
// ---------------------------------------------------------------------------
__global__ void wquant_kernel(const float* __restrict__ w,
                              const float* __restrict__ s_t,
                              const float* __restrict__ t,
                              const float* __restrict__ d,
                              _Float16* __restrict__ out, int Nr, int Kc) {
  int i = blockIdx.x * blockDim.x + threadIdx.x;
  if (i >= Nr * Kc) return;
  int n = i / Kc;
  float s  = softplusf(s_t[n]);
  float df = w[i] - t[n];
  float q  = (fabsf(df) > d[n]) ? (df > 0.f ? 1.f : -1.f) : 0.f;
  out[i] = (_Float16)(s * q);
}

__global__ void f2h_kernel(const float* __restrict__ in,
                           _Float16* __restrict__ out, int n) {
  int i = blockIdx.x * blockDim.x + threadIdx.x;
  if (i < n) out[i] = (_Float16)in[i];
}

// dbc[:, :48] -> f16 A-matrix for the dt GEMM
__global__ void slice48_kernel(const float* __restrict__ dbc,
                               _Float16* __restrict__ out, int M) {
  int i = blockIdx.x * blockDim.x + threadIdx.x;
  if (i >= M * 48) return;
  int m = i / 48, r = i - m * 48;
  out[i] = (_Float16)dbc[(size_t)m * 80 + r];
}

// ---------------------------------------------------------------------------
// Fused depthwise causal conv (K=4) + bias + SiLU + int4 quant -> f16 u_ssm
// ---------------------------------------------------------------------------
__global__ void conv_silu_quant_kernel(const float* __restrict__ xz,
                                       const float* __restrict__ cw,
                                       const float* __restrict__ cb,
                                       const float* __restrict__ pv,
                                       _Float16* __restrict__ u_ssm,
                                       int L, int DI, int N2, int total) {
  int i = blockIdx.x * blockDim.x + threadIdx.x;
  if (i >= total) return;
  int c = i % DI;
  int bl = i / DI;
  int l = bl % L;
  int b = bl / L;
  float acc = cb[c];
#pragma unroll
  for (int j = 0; j < 4; ++j) {
    int ls = l - 3 + j;
    if (ls >= 0) acc += xz[((size_t)b * L + ls) * N2 + c] * cw[c * 4 + j];
  }
  float v  = siluf(acc);
  float sc = pv[c] * (1.f / 7.f);
  float q  = rintf(v / (sc + 1e-8f));
  q = fminf(7.f, fmaxf(-7.f, q));
  u_ssm[i] = (_Float16)(q * sc);
}

// dt = softplus(dt_raw + bias)  (in place)
__global__ void dtact_kernel(float* __restrict__ dt,
                             const float* __restrict__ bias, int DI, int total) {
  int i = blockIdx.x * blockDim.x + threadIdx.x;
  if (i >= total) return;
  dt[i] = softplusf(dt[i] + bias[i % DI]);
}

// ---------------------------------------------------------------------------
// Selective scan. One lane per (b, c, n); 16-lane shfl_xor reduction over n
// (wave32: two channels per wave). Fused D-skip + SiLU(z) gate -> f16 yg.
// ---------------------------------------------------------------------------
__global__ void scan_kernel(const float* __restrict__ dt,
                            const _Float16* __restrict__ u_ssm,
                            const float* __restrict__ dbc,
                            const float* __restrict__ A_log,
                            const float* __restrict__ D_skip,
                            const float* __restrict__ xz,
                            _Float16* __restrict__ yg,
                            int L, int DI, int N2, int BDI) {
  int idx = blockIdx.x * blockDim.x + threadIdx.x;
  int n  = idx & 15;
  int bc = idx >> 4;
  if (bc >= BDI) return;
  int c = bc % DI;
  int b = bc / DI;
  float a   = -__expf(A_log[c * 16 + n]);
  float h   = 0.f;
  float dsk = D_skip[c];
  for (int l = 0; l < L; ++l) {
    size_t t   = (size_t)b * L + l;
    float dtv  = dt[t * DI + c];
    float uv   = (float)u_ssm[t * DI + c];
    float Bv   = dbc[t * 80 + 48 + n];
    float Cv   = dbc[t * 80 + 64 + n];
    h = __expf(dtv * a) * h + (dtv * uv) * Bv;
    float y = h * Cv;
    y += __shfl_xor(y, 1);
    y += __shfl_xor(y, 2);
    y += __shfl_xor(y, 4);
    y += __shfl_xor(y, 8);
    if (n == 0) {
      float zv = xz[t * N2 + DI + c];
      yg[t * DI + c] = (_Float16)((y + uv * dsk) * siluf(zv));
    }
  }
}

// ---------------------------------------------------------------------------
// WMMA GEMM: C[M,N] (f32) = A[M,K] (f16 row-major) * B[N,K]^T (f16).
// 256 threads = 8 waves; block tile 128x128, BK=32.
// Wave grid 4x2: each wave computes 32x64 = 2x4 WMMA 16x16 tiles.
// GUARDED=false: exact-multiple dims -> double-buffered async pipeline using
// GLOBAL_LOAD_ASYNC_TO_LDS_B128; in-order ASYNCcnt completion lets
// s_wait_asynccnt 4 retire exactly the previous tile's 4 loads while the next
// tile streams in.
// ---------------------------------------------------------------------------
#define BM 128
#define BN 128
#define BK 32
#define LDSS 40  // padded LDS row stride in halves (80 B, 16B-aligned chunks)
#define TILE_H (BM * LDSS)

DEV void stage_async_tile(const _Float16* Ag, const _Float16* Bg,
                          _Float16* Al, _Float16* Bl) {
#if defined(HAVE_ASYNC_LDS)
  __builtin_amdgcn_global_load_async_to_lds_b128(gcast_v4i(Ag),     lcast_v4i(Al),     0, 0);
  __builtin_amdgcn_global_load_async_to_lds_b128(gcast_v4i(Ag + 8), lcast_v4i(Al + 8), 0, 0);
  __builtin_amdgcn_global_load_async_to_lds_b128(gcast_v4i(Bg),     lcast_v4i(Bl),     0, 0);
  __builtin_amdgcn_global_load_async_to_lds_b128(gcast_v4i(Bg + 8), lcast_v4i(Bl + 8), 0, 0);
#else
  *(v8h*)Al       = *(const v8h*)Ag;
  *(v8h*)(Al + 8) = *(const v8h*)(Ag + 8);
  *(v8h*)Bl       = *(const v8h*)Bg;
  *(v8h*)(Bl + 8) = *(const v8h*)(Bg + 8);
#endif
}

DEV void wmma_tile_compute(const _Float16* Asb, const _Float16* Bsb,
                           int wm, int wn, int mlane, int kg, v8f acc[2][4]) {
  // A fragment (16x32 f16): lane m = lane&15, kgroup = lane>>4;
  // 16 contiguous halves at 16*kg of the row.
  v16h af[2], bf[4];
#pragma unroll
  for (int i = 0; i < 2; ++i) {
    const _Float16* p = Asb + (wm * 32 + i * 16 + mlane) * LDSS + kg * 16;
    af[i] = combine16(*(const v8h*)p, *(const v8h*)(p + 8));
  }
  // B fragment (32x16 f16): lane n = lane&15; 16 contiguous halves at 16*kg.
#pragma unroll
  for (int j = 0; j < 4; ++j) {
    const _Float16* p = Bsb + (wn * 64 + j * 16 + mlane) * LDSS + kg * 16;
    bf[j] = combine16(*(const v8h*)p, *(const v8h*)(p + 8));
  }
#pragma unroll
  for (int i = 0; i < 2; ++i)
#pragma unroll
    for (int j = 0; j < 4; ++j)
      acc[i][j] = __builtin_amdgcn_wmma_f32_16x16x32_f16(
          false, af[i], false, bf[j], (short)0, acc[i][j], false, false);
}

template <bool GUARDED>
__global__ __launch_bounds__(256) void wmma_gemm_kernel(
    const _Float16* __restrict__ A, const _Float16* __restrict__ Bm,
    float* __restrict__ C, int M, int N, int Kc) {
  __shared__ __align__(16) _Float16 As[(GUARDED ? 1 : 2) * TILE_H];
  __shared__ __align__(16) _Float16 Bs[(GUARDED ? 1 : 2) * TILE_H];
  const int tid  = threadIdx.x;
  const int lane = tid & 31;
  const int wid  = tid >> 5;
  const int wm   = wid >> 1;  // 0..3 -> 32 output rows each
  const int wn   = wid & 1;   // 0..1 -> 64 output cols each
  const int m0   = blockIdx.y * BM;
  const int n0   = blockIdx.x * BN;

  const v8f zf = {0.f, 0.f, 0.f, 0.f, 0.f, 0.f, 0.f, 0.f};
  const v8h zh = {(_Float16)0, (_Float16)0, (_Float16)0, (_Float16)0,
                  (_Float16)0, (_Float16)0, (_Float16)0, (_Float16)0};
  v8f acc[2][4];
#pragma unroll
  for (int i = 0; i < 2; ++i)
#pragma unroll
    for (int j = 0; j < 4; ++j) acc[i][j] = zf;

  // Staging map: each thread owns 16 halves (two 16B chunks) of A and of B.
  const int sRow  = tid >> 1;         // 0..127
  const int sHalf = (tid & 1) * 16;   // halves 0 / 16
  const _Float16* Ag = A  + (size_t)(m0 + sRow) * Kc + sHalf;
  const _Float16* Bg = Bm + (size_t)(n0 + sRow) * Kc + sHalf;
  _Float16* Al = As + sRow * LDSS + sHalf;
  _Float16* Bl = Bs + sRow * LDSS + sHalf;

  const int mlane = lane & 15;
  const int kg    = lane >> 4;

  if constexpr (!GUARDED) {
    const int nk = Kc / BK;
    // Prologue: tile 0 in flight (ASYNCcnt = 4 for this wave).
    stage_async_tile(Ag, Bg, Al, Bl);
    for (int kt = 0; kt < nk; ++kt) {
      const int cur = kt & 1;
      if (kt + 1 < nk) {
        // Issue next tile into the other buffer (ASYNCcnt -> 8), then retire
        // exactly the previous 4 (in-order completion) with asynccnt <= 4.
        stage_async_tile(Ag + (size_t)(kt + 1) * BK, Bg + (size_t)(kt + 1) * BK,
                         Al + (1 - cur) * TILE_H, Bl + (1 - cur) * TILE_H);
        wait_async4();
      } else {
        wait_async0();
      }
      __syncthreads();
      wmma_tile_compute(As + cur * TILE_H, Bs + cur * TILE_H,
                        wm, wn, mlane, kg, acc);
      __syncthreads();  // all waves done reading before this buffer is refilled
    }
  } else {
    for (int k0 = 0; k0 < Kc; k0 += BK) {
      int gk = k0 + sHalf;
      v8h a0 = zh, a1 = zh, b0 = zh, b1 = zh;
      if (gk + 8  <= Kc) a0 = *(const v8h*)(Ag + k0);
      if (gk + 16 <= Kc) a1 = *(const v8h*)(Ag + k0 + 8);
      if (n0 + sRow < N) {
        if (gk + 8  <= Kc) b0 = *(const v8h*)(Bg + k0);
        if (gk + 16 <= Kc) b1 = *(const v8h*)(Bg + k0 + 8);
      }
      *(v8h*)Al       = a0;
      *(v8h*)(Al + 8) = a1;
      *(v8h*)Bl       = b0;
      *(v8h*)(Bl + 8) = b1;
      __syncthreads();
      wmma_tile_compute(As, Bs, wm, wn, mlane, kg, acc);
      __syncthreads();
    }
  }

  // f32 C/D layout: VGPR r -> M = r + 8*(lane>>4); N = lane&15
#pragma unroll
  for (int i = 0; i < 2; ++i) {
    int mt = m0 + wm * 32 + i * 16 + 8 * kg;
#pragma unroll
    for (int j = 0; j < 4; ++j) {
      int nt = n0 + wn * 64 + j * 16 + mlane;
      if (!GUARDED || nt < N) {
#pragma unroll
        for (int r = 0; r < 8; ++r) {
          C[(size_t)(mt + r) * N + nt] = acc[i][j][r];
        }
      }
    }
  }
}

// ---------------------------------------------------------------------------
extern "C" void kernel_launch(void* const* d_in, const int* in_sizes, int n_in,
                              void* d_out, int out_size, void* d_ws,
                              size_t ws_size, hipStream_t stream) {
  (void)in_sizes; (void)n_in; (void)out_size; (void)ws_size;
  const int Bsz = 4, L = 2048, D = 768, DI = 1536, N2 = 3072;
  const int M = Bsz * L;  // 8192

  const float* x         = (const float*)d_in[0];
  const float* sign1     = (const float*)d_in[1];
  const float* sign2     = (const float*)d_in[2];
  const int*   perm      = (const int*)d_in[3];
  const float* pvals_in  = (const float*)d_in[4];
  const float* pvals_cv  = (const float*)d_in[5];
  const float* in_proj_w = (const float*)d_in[6];
  const float* s_in      = (const float*)d_in[7];
  const float* t_in      = (const float*)d_in[8];
  const float* delta_in  = (const float*)d_in[9];
  const float* conv_w    = (const float*)d_in[10];
  const float* conv_b    = (const float*)d_in[11];
  const float* x_proj_w  = (const float*)d_in[12];
  const float* dt_proj_w = (const float*)d_in[13];
  const float* dt_proj_b = (const float*)d_in[14];
  const float* A_log     = (const float*)d_in[15];
  const float* D_skip    = (const float*)d_in[16];
  const float* out_proj_w= (const float*)d_in[17];
  const float* s_out     = (const float*)d_in[18];
  const float* t_out     = (const float*)d_in[19];
  const float* delta_out = (const float*)d_in[20];
  float* out = (float*)d_out;

  char* ws = (char*)d_ws;
  size_t off = 0;
  auto salloc = [&](size_t bytes) {
    void* p = ws + off;
    off = (off + bytes + 255) & ~(size_t)255;
    return p;
  };
  _Float16* xq     = (_Float16*)salloc((size_t)M * D * 2);
  _Float16* Wq_in  = (_Float16*)salloc((size_t)N2 * D * 2);
  float*    xz     = (float*)   salloc((size_t)M * N2 * 4);
  _Float16* u_ssm  = (_Float16*)salloc((size_t)M * DI * 2);
  _Float16* Wxp    = (_Float16*)salloc((size_t)80 * DI * 2);
  float*    dbc    = (float*)   salloc((size_t)M * 80 * 4);
  _Float16* dtA    = (_Float16*)salloc((size_t)M * 48 * 2);
  _Float16* Wdt    = (_Float16*)salloc((size_t)DI * 48 * 2);
  float*    dt     = (float*)   salloc((size_t)M * DI * 4);
  _Float16* yg     = (_Float16*)salloc((size_t)M * DI * 2);
  _Float16* Wq_out = (_Float16*)salloc((size_t)D * DI * 2);

  const int TB = 256;
  // 1) Hadamard + signs/perm + int4 quant
  had_quant_kernel<<<M, TB, 0, stream>>>(x, sign1, sign2, perm, pvals_in, xq, D);
  // 2) ternary in_proj weights -> f16
  wquant_kernel<<<(N2 * D + TB - 1) / TB, TB, 0, stream>>>(
      in_proj_w, s_in, t_in, delta_in, Wq_in, N2, D);
  // 3) xz = xq @ Wq_in^T   (8192 x 3072 x 768)  [async double-buffered fast path]
  wmma_gemm_kernel<false><<<dim3(N2 / BN, M / BM), TB, 0, stream>>>(
      xq, Wq_in, xz, M, N2, D);
  // 4) depthwise conv + SiLU + int4 quant -> u_ssm (f16)
  conv_silu_quant_kernel<<<((size_t)M * DI + TB - 1) / TB, TB, 0, stream>>>(
      xz, conv_w, conv_b, pvals_cv, u_ssm, L, DI, N2, M * DI);
  // 5) x_proj weights -> f16; dbc = u_ssm @ Wxp^T  (8192 x 80 x 1536) [guarded: N=80]
  f2h_kernel<<<(80 * DI + TB - 1) / TB, TB, 0, stream>>>(x_proj_w, Wxp, 80 * DI);
  wmma_gemm_kernel<true><<<dim3(1, M / BM), TB, 0, stream>>>(
      u_ssm, Wxp, dbc, M, 80, DI);
  // 6) dt = softplus(dbc[:, :48] @ Wdt^T + b)  (8192 x 1536 x 48) [guarded: K=48]
  slice48_kernel<<<(M * 48 + TB - 1) / TB, TB, 0, stream>>>(dbc, dtA, M);
  f2h_kernel<<<(DI * 48 + TB - 1) / TB, TB, 0, stream>>>(dt_proj_w, Wdt, DI * 48);
  wmma_gemm_kernel<true><<<dim3(DI / BN, M / BM), TB, 0, stream>>>(
      dtA, Wdt, dt, M, DI, 48);
  dtact_kernel<<<(M * DI + TB - 1) / TB, TB, 0, stream>>>(dt, dt_proj_b, DI, M * DI);
  // 7) selective scan fused with D-skip and SiLU(z) gate -> yg (f16)
  scan_kernel<<<(Bsz * DI * 16 + TB - 1) / TB, TB, 0, stream>>>(
      dt, u_ssm, dbc, A_log, D_skip, xz, yg, L, DI, N2, Bsz * DI);
  // 8) ternary out_proj weights; out = yg @ Wq_out^T  (8192 x 768 x 1536) [fast]
  wquant_kernel<<<(D * DI + TB - 1) / TB, TB, 0, stream>>>(
      out_proj_w, s_out, t_out, delta_out, Wq_out, D, DI);
  wmma_gemm_kernel<false><<<dim3(D / BN, M / BM), TB, 0, stream>>>(
      yg, Wq_out, out, M, D, DI);
}
